// GPTModel_309237645319
// MI455X (gfx1250) — compile-verified, matrix operand order checked by reading, not run on hardware
//
#include <hip/hip_runtime.h>

// ---------------------------------------------------------------------------
// GPT forward for MI455X (gfx1250).
//
// Roofline: ~750 GFLOP of GEMM vs ~460MB of weights -> HBM time ~20us at
// 23.3TB/s, so this is matrix-core bound. Strategy: one-time fp32->bf16
// conversion of all weights/activations, all GEMMs via
// v_wmma_f32_16x16x32_bf16 with f32 accumulation, fused epilogues, and
// GEMM tile staging through LDS via the Tensor Data Mover
// (tensor_load_to_lds + s_wait_tensorcnt) with TDM row padding matching the
// LDS fragment layout.
// ---------------------------------------------------------------------------

#define D_   1024
#define V_   32000
#define L_   4
#define H_   16
#define HD_  64
#define S_   1024
#define M_   4096
#define B_   4
#define BS_  (B_ * S_)

typedef __attribute__((ext_vector_type(16))) __bf16 v16bf;
typedef __attribute__((ext_vector_type(8)))  float  v8f;
typedef unsigned int u32x4 __attribute__((ext_vector_type(4)));
typedef int          i32x4 __attribute__((ext_vector_type(4)));
typedef int          i32x8 __attribute__((ext_vector_type(8)));

#if __has_builtin(__builtin_amdgcn_tensor_load_to_lds)
#define USE_TDM 1
#else
#define USE_TDM 0
#endif

static __device__ __forceinline__ unsigned short f2bf(float f) {
  unsigned int u = __float_as_uint(f);
  unsigned int r = 0x7FFFu + ((u >> 16) & 1u);   // round-to-nearest-even
  return (unsigned short)((u + r) >> 16);
}

// ------------------------- elementwise kernels -----------------------------

__global__ void k_f32_to_bf16(const float* __restrict__ s,
                              unsigned short* __restrict__ d, long long n) {
  long long i  = (long long)blockIdx.x * 256 + threadIdx.x;
  long long st = (long long)gridDim.x * 256;
  for (; i < n; i += st) d[i] = f2bf(s[i]);
}

// W_O (L,H,D,HD) -> packed bf16 (L, D, H*HD)
__global__ void k_pack_wo(const float* __restrict__ wo,
                          unsigned short* __restrict__ out) {
  long long n  = (long long)L_ * D_ * (H_ * HD_);
  long long i  = (long long)blockIdx.x * 256 + threadIdx.x;
  long long st = (long long)gridDim.x * 256;
  for (; i < n; i += st) {
    int c  = (int)(i % (H_ * HD_));
    long long r = i / (H_ * HD_);
    int dd = (int)(r % D_);
    int l  = (int)(r / D_);
    int h  = c / HD_;
    int d  = c % HD_;
    out[i] = f2bf(wo[(((long long)(l * H_ + h) * D_ + dd) * HD_) + d]);
  }
}

// X[b,s,d] = W_E[d, ids[b,s]]  (fp32 + bf16 copies)
__global__ void k_embed(const int* __restrict__ ids, const float* __restrict__ WE,
                        float* __restrict__ X, unsigned short* __restrict__ Xb) {
  long long n  = (long long)BS_ * D_;
  long long i  = (long long)blockIdx.x * 256 + threadIdx.x;
  long long st = (long long)gridDim.x * 256;
  for (; i < n; i += st) {
    long long bs = i / D_;
    int d = (int)(i % D_);
    float v = WE[(long long)d * V_ + ids[bs]];
    X[i] = v;
    Xb[i] = f2bf(v);
  }
}

// Faithful reproduction of the reference's softmax over the BATCH axis:
// P[h,b,t,s] = softmax_b(mask(Z[h,:,t,s])).  Writes P TRANSPOSED in (s,t)
// so the following A = P^T . V GEMM is a plain row-major NN GEMM.
__global__ void k_softmax_batch(const float* __restrict__ Z,
                                unsigned short* __restrict__ Pt) {
  long long n  = (long long)H_ * S_ * S_;
  long long i  = (long long)blockIdx.x * 256 + threadIdx.x;
  long long st = (long long)gridDim.x * 256;
  for (; i < n; i += st) {
    int s = (int)(i % S_);
    long long r = i / S_;
    int t = (int)(r % S_);
    int h = (int)(r / S_);
    bool keep = (t <= s);            // triu(ones) keep mask
    float z[B_];
#pragma unroll
    for (int b = 0; b < B_; ++b)
      z[b] = keep ? Z[(((long long)(h * B_ + b) * S_ + t) * S_) + s] : -1e20f;
    float mx = z[0];
#pragma unroll
    for (int b = 1; b < B_; ++b) mx = fmaxf(mx, z[b]);
    float sum = 0.f;
#pragma unroll
    for (int b = 0; b < B_; ++b) { z[b] = __expf(z[b] - mx); sum += z[b]; }
    float inv = 1.f / sum;
#pragma unroll
    for (int b = 0; b < B_; ++b)
      Pt[(((long long)(h * B_ + b) * S_ + s) * S_) + t] = f2bf(z[b] * inv);
  }
}

// --------------------------- TDM tile loader -------------------------------
// 2D tile (tile_h rows x tile_w elements of 2-byte data) from a row-major
// matrix with row stride `stride_elems` into LDS at byte offset lds_addr.
// TDM padding: pad_interval=3 (every 16 dwords = one 32-half row) and
// pad_amount=3 (4 dwords = 8 halves) -> LDS row pitch of 40 halves, matching
// the fragment loader below.  D# bit layout per CDNA5 ISA ch.8.

#if USE_TDM
static __device__ __forceinline__ void tdm_load_2d(unsigned lds_addr,
                                                   const void* gptr,
                                                   int tile_w, int tile_h,
                                                   int stride_elems) {
  unsigned long long ga = (unsigned long long)gptr;
  u32x4 g0;
  g0.x = 1u;                                             // count=1, user D#
  g0.y = lds_addr;                                       // lds_addr [63:32]
  g0.z = (unsigned)(ga & 0xFFFFFFFFu);                   // global_addr lo
  g0.w = (unsigned)((ga >> 32) & 0x01FFFFFFu) | (2u << 30);  // addr hi|type=2
  i32x8 g1;
  // word0: wg_mask=0 | data_size=1(2B)<<16 | pad_en<<20 | pad_int=3<<22
  //        | pad_amt=3<<25
  g1[0] = (1 << 16) | (1 << 20) | (3 << 22) | (3 << 25);
  // tensor_dim0 = tile_w (bits 79:48), tensor_dim1 = tile_h (bits 111:80)
  g1[1] = (tile_w & 0xFFFF) << 16;
  g1[2] = ((unsigned)tile_w >> 16) | ((tile_h & 0xFFFF) << 16);
  // tile_dim0 (bits 127:112), tile_dim1 (143:128), tile_dim2=0 (159:144)
  g1[3] = ((unsigned)tile_h >> 16) | ((tile_w & 0xFFFF) << 16);
  g1[4] = (tile_h & 0xFFFF);
  // tensor_dim0_stride (bits 207:160); dim1 stride unused for 2D tiles
  g1[5] = stride_elems;
  g1[6] = 0;
  g1[7] = 0;
  i32x4 z4 = {0, 0, 0, 0};
#if defined(__clang_major__) && (__clang_major__ >= 23)
  i32x8 z8 = {0, 0, 0, 0, 0, 0, 0, 0};
  __builtin_amdgcn_tensor_load_to_lds(g0, g1, z4, z4, z8, 0);
#else
  __builtin_amdgcn_tensor_load_to_lds(g0, g1, z4, z4, 0);
#endif
}
#endif

// ------------------------------ WMMA GEMM ----------------------------------
// Block tile 128(M) x 64(N) x 32(K); 8 waves, each wave computes a 32x32
// C tile as 2x2 v_wmma_f32_16x16x32_bf16.  LDS rows pitched at 40 halves.
//
// EPI: 0 = store f32, 1 = add into f32 (residual), 2 = bf16(relu(x)^2),
//      3 = store bf16.   BT: true -> C = A . B^T, false -> C = A . B.
// Batching: z = blockIdx.z decomposed as (h = z/bmod, b = z%bmod) with
// independent element strides per operand.

#define LDSW 40

static __device__ __forceinline__ v16bf frag_load(const unsigned short* row0,
                                                  int rowBase, int lane) {
  // 16-bit operand layout (ISA 7.12.2): lanes 0-15 row M=lane, K = 0..7 and
  // 16..23 ; lanes 16-31 same rows, K offset +8.
  union { v16bf v; unsigned int u[8]; } f;
  const unsigned short* p = row0 + (rowBase + (lane & 15)) * LDSW;
  int kb = (lane >> 4) * 8;
#pragma unroll
  for (int j = 0; j < 8; ++j) {
    int k = kb + ((j & 3) << 1) + ((j >> 2) << 4);
    f.u[j] = *reinterpret_cast<const unsigned int*>(p + k);
  }
  return f.v;
}

template <int EPI, bool BT>
__global__ __launch_bounds__(256, 2) void k_gemm(
    const unsigned short* A, const unsigned short* Bm,
    float* __restrict__ C, unsigned short* __restrict__ Cb,
    int lda, int ldb, int ldc, int Kdim,
    long long sAh, long long sAb, long long sBh, long long sBb,
    long long sCh, long long sCb, int bmod) {
  __shared__ unsigned short lA[128 * LDSW];
  __shared__ unsigned short lB[64 * LDSW];

  int z = blockIdx.z;
  int bh = bmod ? (z / bmod) : 0;
  int bb = bmod ? (z % bmod) : 0;
  A  += (long long)bh * sAh + (long long)bb * sAb;
  Bm += (long long)bh * sBh + (long long)bb * sBb;
  long long cOff = (long long)bh * sCh + (long long)bb * sCb;

  int tid  = threadIdx.x;
  int lane = tid & 31;
  int wave = tid >> 5;
  int wm = (wave >> 1) * 32;   // 4 waves along M
  int wn = (wave & 1) * 32;    // 2 waves along N
  long long mBase = (long long)blockIdx.x * 128;
  long long nBase = (long long)blockIdx.y * 64;

  // cooperative-copy indices (fallback / NN path)
  int kk = tid >> 3, ng = (tid & 7) * 8; // B(NN): 32 k-rows x 8 groups of 8
#if !USE_TDM
  int arow = tid >> 1, aseg = tid & 1;   // A: 128 rows x 2 segs of 16 halves
  int brow = tid >> 2, bseg = tid & 3;   // B(NT): 64 rows x 4 segs of 8 halves
#endif
#if USE_TDM
  unsigned ldsA = (unsigned)(size_t)&lA[0];
  unsigned ldsB = (unsigned)(size_t)&lB[0];
#endif

  v8f acc[2][2] = {};

  int kTiles = Kdim >> 5;
  for (int kt = 0; kt < kTiles; ++kt) {
    __syncthreads();   // previous compute done before tiles are overwritten
#if USE_TDM
    if (wave == 0) {
      tdm_load_2d(ldsA, A + mBase * lda + kt * 32, 32, 128, lda);
      if (BT)
        tdm_load_2d(ldsB, Bm + nBase * ldb + kt * 32, 32, 64, ldb);
      __builtin_amdgcn_s_wait_tensorcnt(0);
    }
#else
    {
      const uint4* g = reinterpret_cast<const uint4*>(
          A + (mBase + arow) * lda + kt * 32 + aseg * 16);
      uint4* sd = reinterpret_cast<uint4*>(&lA[arow * LDSW + aseg * 16]);
      sd[0] = g[0];
      sd[1] = g[1];
    }
    if (BT) {
      const uint4* g = reinterpret_cast<const uint4*>(
          Bm + (nBase + brow) * ldb + kt * 32 + bseg * 8);
      uint4* sd = reinterpret_cast<uint4*>(&lB[brow * LDSW + bseg * 8]);
      sd[0] = g[0];
    }
#endif
    if (!BT) {
      // B row-major KxN: transpose into LDS [n][k]
      int kclamp = (kt + 1 < kTiles) ? (kt + 1) : kt;
      __builtin_prefetch(Bm + (long long)(kclamp * 32 + kk) * ldb + nBase + ng,
                         0, 3);
      uint4 v = *reinterpret_cast<const uint4*>(
          Bm + (long long)(kt * 32 + kk) * ldb + nBase + ng);
      const unsigned short* hv = reinterpret_cast<const unsigned short*>(&v);
#pragma unroll
      for (int q = 0; q < 8; ++q) lB[(ng + q) * LDSW + kk] = hv[q];
    }
    __syncthreads();

    v16bf a0 = frag_load(lA, wm, lane);
    v16bf a1 = frag_load(lA, wm + 16, lane);
    v16bf b0 = frag_load(lB, wn, lane);
    v16bf b1 = frag_load(lB, wn + 16, lane);

    acc[0][0] = __builtin_amdgcn_wmma_f32_16x16x32_bf16(
        false, a0, false, b0, (short)0, acc[0][0], false, false);
    acc[0][1] = __builtin_amdgcn_wmma_f32_16x16x32_bf16(
        false, a0, false, b1, (short)0, acc[0][1], false, false);
    acc[1][0] = __builtin_amdgcn_wmma_f32_16x16x32_bf16(
        false, a1, false, b0, (short)0, acc[1][0], false, false);
    acc[1][1] = __builtin_amdgcn_wmma_f32_16x16x32_bf16(
        false, a1, false, b1, (short)0, acc[1][1], false, false);
  }

  // Epilogue: C/D VGPR layout -> lanes 0-15: M=v, lanes 16-31: M=v+8; N=lane&15
  int r = lane & 15, hi = lane >> 4;
#pragma unroll
  for (int i = 0; i < 2; ++i)
#pragma unroll
    for (int j = 0; j < 2; ++j)
#pragma unroll
      for (int v = 0; v < 8; ++v) {
        long long m = mBase + wm + i * 16 + hi * 8 + v;
        long long n = nBase + wn + j * 16 + r;
        long long idx = cOff + m * ldc + n;
        float x = acc[i][j][v];
        if (EPI == 0) C[idx] = x;
        else if (EPI == 1) C[idx] += x;
        else if (EPI == 2) { float t = x > 0.f ? x : 0.f; Cb[idx] = f2bf(t * t); }
        else Cb[idx] = f2bf(x);
      }
}

// ------------------------------- host side ---------------------------------

extern "C" void kernel_launch(void* const* d_in, const int* in_sizes, int n_in,
                              void* d_out, int out_size, void* d_ws, size_t ws_size,
                              hipStream_t stream) {
  (void)in_sizes; (void)n_in; (void)out_size; (void)ws_size;
  const int*   ids  = (const int*)d_in[0];
  const float* WE   = (const float*)d_in[1];
  const float* WQKV = (const float*)d_in[2];
  const float* WO   = (const float*)d_in[3];
  const float* WFC1 = (const float*)d_in[4];
  const float* WFC2 = (const float*)d_in[5];
  const float* WU   = (const float*)d_in[6];
  float* out = (float*)d_out;

  // bump allocator on workspace (~660 MB total)
  char* wsp = (char*)d_ws;
  size_t off = 0;
  auto alloc = [&](size_t bytes) -> void* {
    void* p = wsp + off;
    off = (off + bytes + 255) & ~(size_t)255;
    return p;
  };
  float*          Xf    = (float*)alloc((size_t)BS_ * D_ * 4);
  unsigned short* Xb    = (unsigned short*)alloc((size_t)BS_ * D_ * 2);
  unsigned short* QKVb  = (unsigned short*)alloc((size_t)BS_ * 3 * D_ * 2);
  float*          Zf    = (float*)alloc((size_t)H_ * B_ * S_ * S_ * 4);
  unsigned short* Ptb   = (unsigned short*)alloc((size_t)H_ * B_ * S_ * S_ * 2);
  unsigned short* Attb  = (unsigned short*)alloc((size_t)BS_ * D_ * 2);
  unsigned short* H1b   = (unsigned short*)alloc((size_t)BS_ * M_ * 2);
  unsigned short* Wqkvb = (unsigned short*)alloc((size_t)L_ * 3 * D_ * D_ * 2);
  unsigned short* Wob   = (unsigned short*)alloc((size_t)L_ * D_ * D_ * 2);
  unsigned short* Wfc1b = (unsigned short*)alloc((size_t)L_ * M_ * D_ * 2);
  unsigned short* Wfc2b = (unsigned short*)alloc((size_t)L_ * D_ * M_ * 2);
  unsigned short* Wub   = (unsigned short*)alloc((size_t)V_ * D_ * 2);

  const dim3 blk(256);
  const dim3 cg(8192);

  // weight conversions (re-done every call: deterministic)
  k_f32_to_bf16<<<cg, blk, 0, stream>>>(WQKV, Wqkvb, (long long)L_ * 3 * D_ * D_);
  k_pack_wo   <<<cg, blk, 0, stream>>>(WO, Wob);
  k_f32_to_bf16<<<cg, blk, 0, stream>>>(WFC1, Wfc1b, (long long)L_ * M_ * D_);
  k_f32_to_bf16<<<cg, blk, 0, stream>>>(WFC2, Wfc2b, (long long)L_ * D_ * M_);
  k_f32_to_bf16<<<cg, blk, 0, stream>>>(WU,   Wub,   (long long)V_ * D_);

  k_embed<<<cg, blk, 0, stream>>>(ids, WE, Xf, Xb);

  for (int l = 0; l < L_; ++l) {
    const unsigned short* wq = Wqkvb + (size_t)l * 3 * D_ * D_;

    // QKV: [4096 x 1024] x [3072 x 1024]^T -> bf16 [4096 x 3072]
    k_gemm<3, true><<<dim3(BS_ / 128, (3 * D_) / 64, 1), blk, 0, stream>>>(
        Xb, wq, nullptr, QKVb, D_, D_, 3 * D_, D_,
        0, 0, 0, 0, 0, 0, 0);

    // Scores Z[h,b,t,s] = K . Q^T over hd, batched over (h,b)
    k_gemm<0, true><<<dim3(S_ / 128, S_ / 64, H_ * B_), blk, 0, stream>>>(
        QKVb + (size_t)H_ * HD_,  /* K slice (k=1) */
        QKVb,                      /* Q slice (k=0) */
        Zf, nullptr, 3 * D_, 3 * D_, S_, HD_,
        /*sAh*/ 64, /*sAb*/ (long long)S_ * 3 * D_,
        /*sBh*/ 64, /*sBb*/ (long long)S_ * 3 * D_,
        /*sCh*/ (long long)B_ * S_ * S_, /*sCb*/ (long long)S_ * S_, B_);

    // batch-axis softmax + transpose -> Pt[h,b,s,t] (bf16)
    k_softmax_batch<<<cg, blk, 0, stream>>>(Zf, Ptb);

    // A[h,b,s,d] = Pt . V  (NN GEMM), output packed as [b,s,(h,d)] bf16
    k_gemm<3, false><<<dim3(S_ / 128, HD_ / 64, H_ * B_), blk, 0, stream>>>(
        Ptb, QKVb + (size_t)2 * H_ * HD_, nullptr, Attb,
        S_, 3 * D_, D_, S_,
        /*sAh*/ (long long)B_ * S_ * S_, /*sAb*/ (long long)S_ * S_,
        /*sBh*/ 64, /*sBb*/ (long long)S_ * 3 * D_,
        /*sCh*/ 64, /*sCb*/ (long long)S_ * D_, B_);

    // X += Att . Wo^T  (residual add into fp32 X)
    k_gemm<1, true><<<dim3(BS_ / 128, D_ / 64, 1), blk, 0, stream>>>(
        Attb, Wob + (size_t)l * D_ * D_, Xf, nullptr, D_, D_, D_, D_,
        0, 0, 0, 0, 0, 0, 0);
    k_f32_to_bf16<<<cg, blk, 0, stream>>>(Xf, Xb, (long long)BS_ * D_);

    // H1 = relu(X . Wfc1^T)^2  (fused squared-relu -> bf16)
    k_gemm<2, true><<<dim3(BS_ / 128, M_ / 64, 1), blk, 0, stream>>>(
        Xb, Wfc1b + (size_t)l * M_ * D_, nullptr, H1b, D_, D_, M_, D_,
        0, 0, 0, 0, 0, 0, 0);

    // X += H1 . Wfc2^T
    k_gemm<1, true><<<dim3(BS_ / 128, D_ / 64, 1), blk, 0, stream>>>(
        H1b, Wfc2b + (size_t)l * D_ * M_, Xf, nullptr, M_, M_, D_, M_,
        0, 0, 0, 0, 0, 0, 0);
    k_f32_to_bf16<<<cg, blk, 0, stream>>>(Xf, Xb, (long long)BS_ * D_);
  }

  // logits = X . Wu^T -> fp32 (B,S,V)
  k_gemm<0, true><<<dim3(BS_ / 128, V_ / 64, 1), blk, 0, stream>>>(
      Xb, Wub, out, nullptr, D_, D_, V_, D_,
      0, 0, 0, 0, 0, 0, 0);
}
